// MultiViewFeatureExtractor_29910152249795
// MI455X (gfx1250) — matrix operand order repaired and checked by reading, not compiled, not run on hardware
//
#include <hip/hip_runtime.h>
#include <hip/hip_bf16.h>

typedef __attribute__((ext_vector_type(16))) _Float16 v16h;
typedef __attribute__((ext_vector_type(8)))  _Float16 v8h;
typedef __attribute__((ext_vector_type(8)))  float    v8f;

#define VV   3
#define NND  8192
#define HHD  128

// ---------------------------------------------------------------------------
// 1. Degree + bitmask pack. One block (8 waves) per (view,row).
//    Fully coalesced single pass over the 805MB int32 adjacency.
//    Wave32 ballot builds one 32-bit mask per chunk; diagonal forced to 1, so
//    deg = sum of popcounts directly (the +1 self-loop is the diag bit).
// ---------------------------------------------------------------------------
__global__ void deg_bits_kernel(const int* __restrict__ adj,
                                float* __restrict__ dinv,
                                unsigned* __restrict__ bits) {
  int b = blockIdx.x;                        // v*8192 + row
  int row = b & (NND - 1);
  const int* arow = adj + ((size_t)b << 13);
  unsigned* brow = bits + ((size_t)b << 8);
  int w = threadIdx.x >> 5, lane = threadIdx.x & 31;
  int cnt = 0;
  for (int c = w; c < 256; c += 8) {
    int col = (c << 5) + lane;
    int a = arow[col];
    int bit = (col == row) ? 1 : (a != 0 ? 1 : 0);
    unsigned mask = (unsigned)__ballot(bit);
    if (lane == 0) { brow[c] = mask; cnt += __popc(mask); }
  }
  __shared__ int red[8];
  if (lane == 0) red[w] = cnt;
  __syncthreads();
  if (threadIdx.x == 0) {
    int deg = 0;
    #pragma unroll
    for (int i = 0; i < 8; ++i) deg += red[i];
    dinv[b] = rsqrtf((float)deg);
  }
}

// ---------------------------------------------------------------------------
// 2. Pack K x Ncols (row-major) into WMMA B-fragment layout via LDS so both
//    the global read and the record write are coalesced.
//    blockIdx.x = K-chunk, blockIdx.y = view (K = gridDim.x*32 per view).
// ---------------------------------------------------------------------------
__global__ void pack_b32_lds(const float* __restrict__ src,
                             const float* __restrict__ rowscale,
                             _Float16* __restrict__ dst, int Ncols) {
  __shared__ _Float16 tile[32 * 256];          // supports Ncols <= 256 (16KB)
  int kch = blockIdx.x;
  int K   = gridDim.x << 5;
  int v   = blockIdx.y;
  src += (size_t)v * K * Ncols;
  if (rowscale) rowscale += (size_t)v * K;
  dst += (size_t)v * K * Ncols;
  int tid = threadIdx.x;
  int nelem = 32 * Ncols;
  for (int i = tid; i < nelem; i += 256) {
    int k = i / Ncols, n = i - k * Ncols;
    int gk = (kch << 5) + k;
    float s = rowscale ? rowscale[gk] : 1.0f;
    tile[i] = (_Float16)(src[(size_t)gk * Ncols + n] * s);
  }
  __syncthreads();
  int nslab = Ncols >> 4;
  int nrec = nslab * 32;
  for (int r = tid; r < nrec; r += 256) {
    int lane = r & 31, slab = r >> 5;
    int n = slab * 16 + (lane & 15);
    int kb = (lane >> 4) << 3;
    v16h o;
    #pragma unroll
    for (int t = 0; t < 16; ++t) o[t] = tile[(kb + t + ((t >> 3) << 3)) * Ncols + n];
    *(v16h*)(dst + (((size_t)kch * nslab + slab) * 32 + lane) * 16) = o;
  }
}

__global__ void pack_b16_lds(const _Float16* __restrict__ src,
                             _Float16* __restrict__ dst, int Ncols) {
  __shared__ _Float16 tile[32 * 256];
  int kch = blockIdx.x;
  int K   = gridDim.x << 5;
  int v   = blockIdx.y;
  src += (size_t)v * K * Ncols;
  dst += (size_t)v * K * Ncols;
  int tid = threadIdx.x;
  int nelem = 32 * Ncols;
  for (int i = tid; i < nelem; i += 256) {
    int k = i / Ncols, n = i - k * Ncols;
    tile[i] = src[(size_t)((kch << 5) + k) * Ncols + n];
  }
  __syncthreads();
  int nslab = Ncols >> 4;
  int nrec = nslab * 32;
  for (int r = tid; r < nrec; r += 256) {
    int lane = r & 31, slab = r >> 5;
    int n = slab * 16 + (lane & 15);
    int kb = (lane >> 4) << 3;
    v16h o;
    #pragma unroll
    for (int t = 0; t < 16; ++t) o[t] = tile[(kb + t + ((t >> 3) << 3)) * Ncols + n];
    *(v16h*)(dst + (((size_t)kch * nslab + slab) * 32 + lane) * 16) = o;
  }
}

// ---------------------------------------------------------------------------
// 3. GCN aggregation: out[n,:] = relu(dinv[n]*(A_bin[n,:] @ Bt) + bias).
//    blockIdx.y = view. M=32 rows/block, 8 waves (one 16-col slab each,
//    2 WMMAs/step), double-buffered LDS A tile (1 barrier/step), bitmask
//    words register-cached 4 steps at a time, select-based bit->f16 pack.
// ---------------------------------------------------------------------------
__global__ void gcn_agg(const unsigned* __restrict__ bits,
                        const float* __restrict__ dinv,
                        const _Float16* __restrict__ Bt,
                        const float* __restrict__ bias,
                        _Float16* __restrict__ outh,
                        float* __restrict__ outf) {
  __shared__ __align__(32) _Float16 Atile[2][32 * 32];
  int v = blockIdx.y;
  bits += (size_t)v * NND * 256;
  dinv += (size_t)v * NND;
  Bt   += (size_t)v * NND * HHD;
  bias += (size_t)v * HHD;
  outh += (size_t)v * NND * HHD;
  if (outf) outf += (size_t)v * NND * HHD;

  int tid  = threadIdx.x;
  int w    = tid >> 5;
  int lane = tid & 31;
  int half = lane >> 4;
  int m    = lane & 15;
  int r0   = blockIdx.x << 5;

  int srow = tid >> 3;               // staging row 0..31
  int spos = (tid & 7) << 2;         // staging nibble start bit

  const uint4* bw = (const uint4*)(bits + ((size_t)(r0 + srow) << 8));
  unsigned mwa[4];
  { uint4 q = bw[0]; mwa[0] = q.x; mwa[1] = q.y; mwa[2] = q.z; mwa[3] = q.w; }

  // stage kc = 0 into buffer 0
  {
    unsigned p = (mwa[0] >> spos) & 15u;
    uint2 d;
    d.x = ((p & 1u) ? 0x3C00u : 0u) | ((p & 2u) ? 0x3C000000u : 0u);
    d.y = ((p & 4u) ? 0x3C00u : 0u) | ((p & 8u) ? 0x3C000000u : 0u);
    *(uint2*)&Atile[0][srow * 32 + spos] = d;
  }
  __syncthreads();

  v8f acc0 = {}, acc1 = {};
  #pragma unroll 4
  for (int kc = 0; kc < (NND >> 5); ++kc) {
    int nxt = kc + 1;
    if (nxt < (NND >> 5)) {
      if ((nxt & 3) == 0) {
        uint4 q = bw[nxt >> 2];
        mwa[0] = q.x; mwa[1] = q.y; mwa[2] = q.z; mwa[3] = q.w;
      }
      unsigned p = (mwa[nxt & 3] >> spos) & 15u;
      uint2 d;
      d.x = ((p & 1u) ? 0x3C00u : 0u) | ((p & 2u) ? 0x3C000000u : 0u);
      d.y = ((p & 4u) ? 0x3C00u : 0u) | ((p & 8u) ? 0x3C000000u : 0u);
      *(uint2*)&Atile[nxt & 1][srow * 32 + spos] = d;
    }

    int buf = kc & 1;
    v8h lo0 = *(const v8h*)&Atile[buf][m * 32 + 8 * half];
    v8h hi0 = *(const v8h*)&Atile[buf][m * 32 + 16 + 8 * half];
    v8h lo1 = *(const v8h*)&Atile[buf][(m + 16) * 32 + 8 * half];
    v8h hi1 = *(const v8h*)&Atile[buf][(m + 16) * 32 + 16 + 8 * half];
    v16h a0, a1;
    #pragma unroll
    for (int i = 0; i < 8; ++i) {
      a0[i] = lo0[i]; a0[i + 8] = hi0[i];
      a1[i] = lo1[i]; a1[i + 8] = hi1[i];
    }
    v16h bfr = *(const v16h*)(Bt + ((size_t)(kc * 8 + w) * 32 + lane) * 16);

    acc0 = __builtin_amdgcn_wmma_f32_16x16x32_f16(false, a0, false, bfr,
                                                  (short)0, acc0, false, false);
    acc1 = __builtin_amdgcn_wmma_f32_16x16x32_f16(false, a1, false, bfr,
                                                  (short)0, acc1, false, false);
    __syncthreads();
  }

  int col = w * 16 + m;
  #pragma unroll
  for (int q = 0; q < 8; ++q) {
    int row = r0 + q + 8 * half;
    float val = dinv[row] * acc0[q] + bias[col];
    val = val > 0.0f ? val : 0.0f;
    outh[(size_t)row * HHD + col] = (_Float16)val;
    if (outf) outf[(size_t)row * HHD + col] = val;
  }
  #pragma unroll
  for (int q = 0; q < 8; ++q) {
    int row = r0 + 16 + q + 8 * half;
    float val = dinv[row] * acc1[q] + bias[col];
    val = val > 0.0f ? val : 0.0f;
    outh[(size_t)row * HHD + col] = (_Float16)val;
    if (outf) outf[(size_t)row * HHD + col] = val;
  }
}

// ---------------------------------------------------------------------------
// 4. Generic f16 WMMA GEMM: out = op(A(row-major) @ Bt), blockDim = 32*nslab.
//    blockIdx.y = view; per-view strides passed explicitly (0 => shared).
// ---------------------------------------------------------------------------
__global__ void gemm_f16(const _Float16* __restrict__ A,
                         const _Float16* __restrict__ Bt,
                         const float* __restrict__ bias,
                         const float* __restrict__ rowscale,
                         _Float16* __restrict__ outh,
                         float* __restrict__ outf,
                         int K, int Ncols, int relu,
                         size_t strA, size_t strB, size_t strRS, size_t strOut) {
  int v = blockIdx.y;
  A  += (size_t)v * strA;
  Bt += (size_t)v * strB;
  if (rowscale) rowscale += (size_t)v * strRS;
  if (outh) outh += (size_t)v * strOut;
  if (outf) outf += (size_t)v * strOut;

  int nslab = Ncols >> 4;
  int tid  = threadIdx.x;
  int w    = tid >> 5;
  int lane = tid & 31;
  int half = lane >> 4;
  int m    = lane & 15;
  int r0   = blockIdx.x << 4;
  int nkc  = K >> 5;

  v8f acc = {};
  #pragma unroll 2
  for (int kc = 0; kc < nkc; ++kc) {
    const _Float16* ap = A + (size_t)(r0 + m) * K + kc * 32 + 8 * half;
    v8h lo = *(const v8h*)ap;
    v8h hi = *(const v8h*)(ap + 16);
    v16h a;
    #pragma unroll
    for (int i = 0; i < 8; ++i) { a[i] = lo[i]; a[i + 8] = hi[i]; }
    v16h b = *(const v16h*)(Bt + ((size_t)(kc * nslab + w) * 32 + lane) * 16);
    acc = __builtin_amdgcn_wmma_f32_16x16x32_f16(false, a, false, b,
                                                 (short)0, acc, false, false);
  }

  int col = w * 16 + m;
  #pragma unroll
  for (int q = 0; q < 8; ++q) {
    int row = r0 + q + 8 * half;
    float val = acc[q];
    if (rowscale) val *= rowscale[row];
    if (bias)     val += bias[col];
    if (relu)     val = val > 0.0f ? val : 0.0f;
    if (outh) outh[(size_t)row * Ncols + col] = (_Float16)val;
    if (outf) outf[(size_t)row * Ncols + col] = val;
  }
}

// ---------------------------------------------------------------------------
// 5. Column means of h3 (f32, in d_out h-region): one block per (v,h).
// ---------------------------------------------------------------------------
__global__ void summ_kernel(const float* __restrict__ h3f,
                            float* __restrict__ summ) {
  int v = blockIdx.x >> 7;
  int h = blockIdx.x & 127;
  const float* base = h3f + (size_t)v * NND * HHD + h;
  float s = 0.0f;
  for (int n = threadIdx.x; n < NND; n += 256) s += base[(size_t)n * HHD];
  __shared__ float red[256];
  red[threadIdx.x] = s;
  __syncthreads();
  for (int off = 128; off > 0; off >>= 1) {
    if (threadIdx.x < off) red[threadIdx.x] += red[threadIdx.x + off];
    __syncthreads();
  }
  if (threadIdx.x == 0) summ[blockIdx.x] = red[0] * (1.0f / (float)NND);
}

// ---------------------------------------------------------------------------
// 6. Attention MLP + softmax over V=3. One block of 64 threads.
// ---------------------------------------------------------------------------
__global__ void attn_kernel(const float* __restrict__ summ,
                            const float* __restrict__ Wa1,
                            const float* __restrict__ ba1,
                            const float* __restrict__ Wa2,
                            const float* __restrict__ ba2,
                            float* __restrict__ attn_ws,
                            float* __restrict__ attn_out) {
  __shared__ float red[64];
  __shared__ float sc[VV];
  int t = threadIdx.x;
  for (int v = 0; v < VV; ++v) {
    float acc = ba1[t];
    for (int k = 0; k < HHD; ++k) acc += summ[v * HHD + k] * Wa1[k * 64 + t];
    red[t] = tanhf(acc) * Wa2[t];
    __syncthreads();
    for (int off = 32; off > 0; off >>= 1) {
      if (t < off) red[t] += red[t + off];
      __syncthreads();
    }
    if (t == 0) sc[v] = red[0] + ba2[0];
    __syncthreads();
  }
  if (t == 0) {
    float mx = fmaxf(sc[0], fmaxf(sc[1], sc[2]));
    float e0 = __expf(sc[0] - mx), e1 = __expf(sc[1] - mx), e2 = __expf(sc[2] - mx);
    float inv = 1.0f / (e0 + e1 + e2);
    attn_ws[0] = e0 * inv; attn_ws[1] = e1 * inv; attn_ws[2] = e2 * inv;
    attn_out[0] = e0 * inv; attn_out[1] = e1 * inv; attn_out[2] = e2 * inv;
  }
}

// ---------------------------------------------------------------------------
// 7. Fusion input: fin[n, v*128+h] = attn[v] * h3[v,n,h]   (f16, N x 384)
// ---------------------------------------------------------------------------
__global__ void fin_kernel(const _Float16* __restrict__ h3h,
                           const float* __restrict__ attn,
                           _Float16* __restrict__ fin) {
  int id = blockIdx.x * 256 + threadIdx.x;    // over 8192*384
  int n = id / (VV * HHD);
  int c = id % (VV * HHD);
  int v = c >> 7;
  int h = c & 127;
  fin[id] = (_Float16)(attn[v] * (float)h3h[((size_t)v * NND + n) * HHD + h]);
}

// ---------------------------------------------------------------------------
extern "C" void kernel_launch(void* const* d_in, const int* in_sizes, int n_in,
                              void* d_out, int out_size, void* d_ws, size_t ws_size,
                              hipStream_t stream) {
  (void)in_sizes; (void)n_in; (void)out_size; (void)ws_size;
  const int*   adj = (const int*)d_in[0];
  const float* W1  = (const float*)d_in[1];
  const float* b1  = (const float*)d_in[2];
  const float* W2  = (const float*)d_in[3];
  const float* b2  = (const float*)d_in[4];
  const float* Wa1 = (const float*)d_in[5];
  const float* ba1 = (const float*)d_in[6];
  const float* Wa2 = (const float*)d_in[7];
  const float* ba2 = (const float*)d_in[8];
  const float* Wf1 = (const float*)d_in[9];
  const float* bf1 = (const float*)d_in[10];
  const float* Wf2 = (const float*)d_in[11];
  const float* bf2 = (const float*)d_in[12];
  float* out = (float*)d_out;
  char*  ws  = (char*)d_ws;

  // Workspace layout (bytes, 256-aligned)
  float*    dinv  = (float*)(ws + 0);                 //  98,304
  unsigned* bits  = (unsigned*)(ws + 98304);          //  25,165,824 (L2-resident)
  _Float16* btbig = (_Float16*)(ws + 25264128);       //  6,291,456 : W1bt -> h2bt
  _Float16* w2bt  = (_Float16*)(ws + 31555584);       //  98,304
  _Float16* wf1bt = (_Float16*)(ws + 31653888);       //  196,608
  _Float16* wf2bt = (_Float16*)(ws + 31850496);       //  65,536
  _Float16* buf1  = (_Float16*)(ws + 31916032);       //  6,291,456 : h1(f16) -> fin
  _Float16* buf2  = (_Float16*)(ws + 38207488);       //  6,291,456 : h2'(f16) -> g1
  _Float16* h3h   = (_Float16*)(ws + 44498944);       //  6,291,456
  float*    summ  = (float*)(ws + 50790400);          //  1,536
  float*    attnw = (float*)(ws + 50791936);          //  16

  float* fused_out = out;                              // (8192,128) f32
  float* attn_out  = out + (size_t)NND * HHD;          // (3,)
  float* h_out     = attn_out + VV;                    // (3,8192,128) f32

  const size_t sNH = (size_t)NND * HHD;

  // 1) degree + bitmask (single coalesced pass over 805MB adjacency)
  deg_bits_kernel<<<VV * NND, 256, 0, stream>>>(adj, dinv, bits);

  // 2) pack weights into WMMA B layout; W1 rows pre-scaled by dinv_j
  pack_b32_lds<<<dim3(NND / 32, VV), 256, 0, stream>>>(W1, dinv, btbig, HHD);
  pack_b32_lds<<<dim3(HHD / 32, VV), 256, 0, stream>>>(W2, nullptr, w2bt, HHD);
  pack_b32_lds<<<dim3((VV * HHD) / 32, 1), 256, 0, stream>>>(Wf1, nullptr, wf1bt, 2 * HHD);
  pack_b32_lds<<<dim3((2 * HHD) / 32, 1), 256, 0, stream>>>(Wf2, nullptr, wf2bt, HHD);

  // 3) h1 = relu(dinv .* (A_bin @ W1') + b1)      -> buf1 (f16)
  gcn_agg<<<dim3(NND / 32, VV), 256, 0, stream>>>(bits, dinv, btbig, b1, buf1, nullptr);

  // 4) h2' = dinv .* (h1 @ W2)                    -> buf2 (f16)
  gemm_f16<<<dim3(NND / 16, VV), 256, 0, stream>>>(buf1, w2bt, nullptr, dinv,
                                                   buf2, nullptr, HHD, HHD, 0,
                                                   sNH, (size_t)HHD * HHD,
                                                   (size_t)NND, sNH);

  // 5) repack h2' into B layout (reuses btbig)
  pack_b16_lds<<<dim3(NND / 32, VV), 256, 0, stream>>>(buf2, btbig, HHD);

  // 6) h3 = relu(dinv .* (A_bin @ h2') + b2)      -> h3h (f16) + d_out h (f32)
  gcn_agg<<<dim3(NND / 32, VV), 256, 0, stream>>>(bits, dinv, btbig, b2, h3h, h_out);

  // 7) view summaries + attention softmax
  summ_kernel<<<VV * HHD, 256, 0, stream>>>(h_out, summ);
  attn_kernel<<<1, 64, 0, stream>>>(summ, Wa1, ba1, Wa2, ba2, attnw, attn_out);

  // 8) fusion input (attn-weighted, N x 384)      -> buf1 (f16)
  fin_kernel<<<(NND * VV * HHD) / 256, 256, 0, stream>>>(h3h, attnw, buf1);

  // 9) g1 = relu(fin @ Wf1 + bf1)                 -> buf2 (f16), 16 waves/block
  gemm_f16<<<dim3(NND / 16, 1), 512, 0, stream>>>(buf1, wf1bt, bf1, nullptr,
                                                  buf2, nullptr, VV * HHD, 2 * HHD, 1,
                                                  0, 0, 0, 0);

  // 10) fused = g1 @ Wf2 + bf2                    -> d_out (f32)
  gemm_f16<<<dim3(NND / 16, 1), 256, 0, stream>>>(buf2, wf2bt, bf2, nullptr,
                                                  nullptr, fused_out, 2 * HHD, HHD, 0,
                                                  0, 0, 0, 0);
}
